// SchNetModel_13254269075582
// MI455X (gfx1250) — matrix operand order repaired and checked by reading, not compiled, not run on hardware
//
#include <hip/hip_runtime.h>

typedef __attribute__((ext_vector_type(16))) _Float16 v16h;
typedef __attribute__((ext_vector_type(8)))  _Float16 v8h;
typedef __attribute__((ext_vector_type(4)))  _Float16 v4h;
typedef __attribute__((ext_vector_type(8)))  float    v8f;

#define BATCH  64
#define NATOM  96
#define FIN    16
#define HDIM   128
#define KRBF   64
#define NLAYER 3
#define BN     (BATCH * NATOM)   // 6144

// ---------------------------------------------------------------------------
// prep: per-batch valid count, per-row bias scale (cnt_b if row valid else 0)
// ---------------------------------------------------------------------------
__global__ void prep_kernel(const int* __restrict__ batch,
                            float* __restrict__ cntrow, float* __restrict__ cntb) {
    __shared__ int red[128];
    const int b = blockIdx.x, j = threadIdx.x;
    const int valid = (j < NATOM) ? (batch[b * NATOM + j] != -1) : 0;
    red[j] = valid;
    __syncthreads();
    for (int s = 64; s > 0; s >>= 1) {
        if (j < s) red[j] += red[j + s];
        __syncthreads();
    }
    const int cnt = red[0];
    if (j < NATOM) cntrow[b * NATOM + j] = valid ? (float)cnt : 0.0f;
    if (j == 0)    cntb[b] = (float)cnt;
}

// ---------------------------------------------------------------------------
// Wcomb[l] = Wrbf[l] @ Wpair[l]   ([64,128] = [64,128]@[128,128]), fp32 exact
// ---------------------------------------------------------------------------
__global__ void wcomb_kernel(const float* __restrict__ Wrbf,
                             const float* __restrict__ Wpair,
                             float* __restrict__ Wcomb) {
    const int idx = blockIdx.x * 256 + threadIdx.x;      // < 3*64*128
    const int l = idx / (KRBF * HDIM);
    const int r = idx % (KRBF * HDIM);
    const int k = r / HDIM, h = r % HDIM;
    const float* wa = Wrbf + (l * KRBF + k) * HDIM;      // Wrbf[l][k][:]
    const float* wb = Wpair + l * HDIM * HDIM + h;       // Wpair[l][:][h]
    float acc = 0.0f;
    for (int m = 0; m < HDIM; ++m) acc = fmaf(wa[m], wb[m * HDIM], acc);
    Wcomb[idx] = acc;
}

// bcomb[l] = brbf[l] @ Wpair[l] + bpair[l]
__global__ void bcomb_kernel(const float* __restrict__ brbf,
                             const float* __restrict__ Wpair,
                             const float* __restrict__ bpair,
                             float* __restrict__ bcomb) {
    const int idx = blockIdx.x * 128 + threadIdx.x;      // < 3*128
    const int l = idx / HDIM, h = idx % HDIM;
    const float* wb = Wpair + l * HDIM * HDIM + h;
    float acc = bpair[idx];
    for (int m = 0; m < HDIM; ++m) acc = fmaf(brbf[l * HDIM + m], wb[m * HDIM], acc);
    bcomb[idx] = acc;
}

// ---------------------------------------------------------------------------
// S[b,j,k] = valid_j * sum_i valid_i * exp(-10*(d_ij - c_k)^2)  (fp32)
// one block per (b,j), 64 threads = one per RBF center, R[b] staged in LDS
// ---------------------------------------------------------------------------
__global__ void rbf_sum_kernel(const float* __restrict__ R,
                               const int* __restrict__ batch,
                               float* __restrict__ S) {
    __shared__ float Rs[NATOM][3];
    __shared__ int   vs[NATOM];
    const int b = blockIdx.x / NATOM;
    const int j = blockIdx.x % NATOM;
    const int k = threadIdx.x;                            // 0..63
    for (int t = k; t < NATOM * 3; t += 64) Rs[t / 3][t % 3] = R[b * NATOM * 3 + t];
    for (int t = k; t < NATOM; t += 64)     vs[t] = (batch[b * NATOM + t] != -1);
    __syncthreads();
    const float xj = Rs[j][0], yj = Rs[j][1], zj = Rs[j][2];
    const float ck = (float)k * (10.0f / 63.0f);          // linspace(0,10,64)
    float acc = 0.0f;
    for (int i = 0; i < NATOM; ++i) {
        if (vs[i]) {
            const float dx = Rs[i][0] - xj, dy = Rs[i][1] - yj, dz = Rs[i][2] - zj;
            const float d2 = dx * dx + dy * dy + dz * dz;
            const float d  = (d2 > 0.0f) ? sqrtf(d2) : 0.0f;   // safe norm
            const float t  = d - ck;
            acc += __expf(-10.0f * t * t);
        }
    }
    S[(b * NATOM + j) * KRBF + k] = vs[j] ? acc : 0.0f;
}

// ---------------------------------------------------------------------------
// Generic WMMA GEMM: C[BN,128] = epilogue(A[BN,KDIM] @ W[KDIM,128])
//   block = 256 thr (8 waves); M-tile = 16 rows; wave w owns N-tile [16w,16w+16)
//   A (16x32) and W (32x128, transposed) staged fp32->f16 via LDS with
//   divergence-free clamped loads; fragments read as aligned ds_load_b128;
//   fp32 accumulation via v_wmma_f32_16x16x32_f16.
// LDS padding: row stride 40 halfs = 80 B -> 16 B aligned rows (b128 legal)
//   and bank stride 20 dwords (gcd(20,64)=4, 16 rows all distinct banks).
// ---------------------------------------------------------------------------
constexpr int EPI_BIAS    = 0;   // C = acc + bias[n]
constexpr int EPI_ROWBIAS = 1;   // C = acc + rowscale[m]*bias[n]
constexpr int EPI_SILU    = 2;   // C = silu(acc + bias[n])
constexpr int EPI_ACCUM   = 3;   // C += acc + bias[n]

template <int KDIM, int EPI>
__global__ __launch_bounds__(256)
void gemm_wmma(const float* __restrict__ A, const float* __restrict__ W,
               const float* __restrict__ bias, const float* __restrict__ rowscale,
               float* __restrict__ C) {
    __shared__ _Float16 As[16][40];     // A tile: 16 rows x 32 k
    __shared__ _Float16 Wt[HDIM][40];   // W tile transposed: 128 n x 32 k
    const int tid  = threadIdx.x;
    const int row0 = blockIdx.x * 16;
    const int wave = tid >> 5;
    const int lane = tid & 31;
    const int col0 = wave * 16;
    const int hi   = lane >> 4;         // 0/1 half-wave
    const int nn   = lane & 15;
    // stage-A mapping: one row, two consecutive k per thread
    const int ar  = tid >> 4;           // 0..15
    const int ak0 = (tid & 15) * 2;     // 0,2,..,30
    // stage-W mapping: one n column, 16 k per thread (4 groups of 4)
    const int wn  = tid & 127;          // 0..127
    const int wk  = (tid >> 7) * 16;    // 0 or 16

    v8f c = {};
    constexpr int KSTEPS = (KDIM + 31) / 32;
    for (int kb = 0; kb < KSTEPS; ++kb) {
        // ---- cooperative A stage (unconditional load: clamp index, zero by predicate)
        {
            const int   kg = kb * 32 + ak0;
            const int   ok = (kg < KDIM);
            const float s  = ok ? 1.0f : 0.0f;
            const float2 v = *(const float2*)&A[(row0 + ar) * KDIM + (ok ? kg : 0)];
            As[ar][ak0 + 0] = (_Float16)(v.x * s);
            As[ar][ak0 + 1] = (_Float16)(v.y * s);
        }
        // ---- cooperative W stage, transposed into Wt[n][k]
        //      loads coalesced over n; k-guards wave-uniform; b64 LDS stores
        #pragma unroll
        for (int g = 0; g < 4; ++g) {
            const int kk0 = wk + g * 4;
            v4h h4;
            #pragma unroll
            for (int i = 0; i < 4; ++i) {
                const int   kg = kb * 32 + kk0 + i;
                const int   ok = (kg < KDIM);
                const float f  = W[(ok ? kg : 0) * HDIM + wn];
                h4[i] = (_Float16)(ok ? f : 0.0f);
            }
            *(v4h*)&Wt[wn][kk0] = h4;
        }
        // prefetch next k-step tiles while this step computes
        if (kb + 1 < KSTEPS) {
            __builtin_prefetch(&A[(row0 + ar) * KDIM + kb * 32 + 32], 0, 0);
            __builtin_prefetch(&W[(kb * 32 + 32 + wk) * HDIM + wn], 0, 0);
        }
        __syncthreads();
        // ---- fragments: aligned 16 B LDS vector loads
        const v8h a_lo = *(const v8h*)&As[nn][hi * 8];            // K = hi*8 + 0..7
        const v8h a_hi = *(const v8h*)&As[nn][16 + hi * 8];       // K = 16 + hi*8 + 0..7
        const v8h b_lo = *(const v8h*)&Wt[col0 + nn][hi * 8];     // K = hi*8 + 0..7
        const v8h b_hi = *(const v8h*)&Wt[col0 + nn][16 + hi * 8];// K = 16 + hi*8 + 0..7
        v16h a, b;
        #pragma unroll
        for (int e = 0; e < 8; ++e) {
            a[e] = a_lo[e]; a[e + 8] = a_hi[e];
            b[e] = b_lo[e]; b[e + 8] = b_hi[e];
        }
        c = __builtin_amdgcn_wmma_f32_16x16x32_f16(false, a, false, b,
                                                   (short)0, c, false, false);
        __syncthreads();
    }
    // epilogue: C/D layout -> c[v] holds (M = hi*8 + v, N = nn) of the 16x16 tile
    #pragma unroll
    for (int v = 0; v < 8; ++v) {
        const int m = row0 + hi * 8 + v;
        const int n = col0 + nn;
        float x = c[v];
        if (EPI == EPI_BIAS)    { x += bias[n]; }
        if (EPI == EPI_ROWBIAS) { x += rowscale[m] * bias[n]; }
        if (EPI == EPI_SILU)    { x += bias[n]; x = x / (1.0f + __expf(-x)); }
        if (EPI == EPI_ACCUM)   { x += bias[n]; x += C[m * HDIM + n]; }
        C[m * HDIM + n] = x;
    }
}

// ---------------------------------------------------------------------------
// out[b] = (sum_j valid_j * (u[b,j,:] . Wo2 + bo2)) / cnt_b
// ---------------------------------------------------------------------------
__global__ void out_kernel(const float* __restrict__ U, const float* __restrict__ Wo2,
                           const float* __restrict__ bo2, const int* __restrict__ batch,
                           const float* __restrict__ cntb, float* __restrict__ out) {
    __shared__ float red[128];
    const int b = blockIdx.x, j = threadIdx.x;
    float o = 0.0f;
    if (j < NATOM && batch[b * NATOM + j] != -1) {
        float acc = bo2[0];
        const float* u = U + (b * NATOM + j) * HDIM;
        for (int h = 0; h < HDIM; ++h) acc = fmaf(u[h], Wo2[h], acc);
        o = acc;
    }
    red[j] = o;
    __syncthreads();
    for (int s = 64; s > 0; s >>= 1) {
        if (j < s) red[j] += red[j + s];
        __syncthreads();
    }
    if (j == 0) out[b] = red[0] / fmaxf(cntb[b], 1.0f);
}

// ---------------------------------------------------------------------------
extern "C" void kernel_launch(void* const* d_in, const int* in_sizes, int n_in,
                              void* d_out, int out_size, void* d_ws, size_t ws_size,
                              hipStream_t stream) {
    const float* X     = (const float*)d_in[0];
    const float* R     = (const float*)d_in[1];
    const int*   batch = (const int*)  d_in[2];
    const float* We    = (const float*)d_in[3];
    const float* be    = (const float*)d_in[4];
    const float* Wrbf  = (const float*)d_in[5];
    const float* brbf  = (const float*)d_in[6];
    const float* Wpair = (const float*)d_in[7];
    const float* bpair = (const float*)d_in[8];
    const float* Wa1   = (const float*)d_in[9];
    const float* ba1   = (const float*)d_in[10];
    const float* Wa2   = (const float*)d_in[11];
    const float* ba2   = (const float*)d_in[12];
    const float* Wo1   = (const float*)d_in[13];
    const float* bo1   = (const float*)d_in[14];
    const float* Wo2   = (const float*)d_in[15];
    const float* bo2   = (const float*)d_in[16];
    float* out = (float*)d_out;

    // workspace layout (floats); total ~10.6 MB
    float* ws     = (float*)d_ws;
    float* S      = ws;                       // [BN, 64]
    float* h      = S      + BN * KRBF;       // [BN, 128]
    float* agg    = h      + BN * HDIM;       // [BN, 128]  (also reused as 'u')
    float* t      = agg    + BN * HDIM;       // [BN, 128]
    float* Wcomb  = t      + BN * HDIM;       // [3, 64, 128]
    float* bcomb  = Wcomb  + NLAYER * KRBF * HDIM;  // [3, 128]
    float* cntrow = bcomb  + NLAYER * HDIM;   // [BN]
    float* cntb   = cntrow + BN;              // [B]

    prep_kernel <<<BATCH, 128, 0, stream>>>(batch, cntrow, cntb);
    wcomb_kernel<<<(NLAYER * KRBF * HDIM) / 256, 256, 0, stream>>>(Wrbf, Wpair, Wcomb);
    bcomb_kernel<<<NLAYER, 128, 0, stream>>>(brbf, Wpair, bpair, bcomb);
    rbf_sum_kernel<<<BN, 64, 0, stream>>>(R, batch, S);

    // h = X @ We + be
    gemm_wmma<FIN, EPI_BIAS><<<BN / 16, 256, 0, stream>>>(X, We, be, nullptr, h);

    for (int l = 0; l < NLAYER; ++l) {
        // agg = S @ Wcomb[l] + cntrow * bcomb[l]
        gemm_wmma<KRBF, EPI_ROWBIAS><<<BN / 16, 256, 0, stream>>>(
            S, Wcomb + l * KRBF * HDIM, bcomb + l * HDIM, cntrow, agg);
        // t = silu(agg @ Wa1[l] + ba1[l])
        gemm_wmma<HDIM, EPI_SILU><<<BN / 16, 256, 0, stream>>>(
            agg, Wa1 + l * HDIM * HDIM, ba1 + l * HDIM, nullptr, t);
        // h += t @ Wa2[l] + ba2[l]
        gemm_wmma<HDIM, EPI_ACCUM><<<BN / 16, 256, 0, stream>>>(
            t, Wa2 + l * HDIM * HDIM, ba2 + l * HDIM, nullptr, h);
    }

    // u = silu(h @ Wo1 + bo1)   (reuse agg buffer)
    gemm_wmma<HDIM, EPI_SILU><<<BN / 16, 256, 0, stream>>>(h, Wo1, bo1, nullptr, agg);
    // out[b] = masked mean of (u @ Wo2 + bo2)
    out_kernel<<<BATCH, 128, 0, stream>>>(agg, Wo2, bo2, batch, cntb, out);
}
